// Block_2705829396819
// MI455X (gfx1250) — matrix operand order repaired and checked by reading, not compiled, not run on hardware
//
#include <hip/hip_runtime.h>
#include <hip/hip_bf16.h>
#include <math.h>

typedef __attribute__((ext_vector_type(16))) _Float16 v16h;
typedef __attribute__((ext_vector_type(8)))  float    v8f;
typedef __attribute__((ext_vector_type(4)))  unsigned int v4u;
typedef __attribute__((ext_vector_type(8)))  int      v8i;
typedef __attribute__((ext_vector_type(4)))  int      v4i;
typedef _Float16 half_t;

#define N_EMBD 1024
#define N_HEAD 16
#define HEAD   64
#define SEQ    2048
#define BATCH  4
#define ROWS   (BATCH * SEQ)   // 8192
#define FFN    4096
#define NEGINF (-1.0e30f)

// ---------------------------------------------------------------------------
// WMMA fragment loaders (gfx1250 wave32 layouts, CDNA5 ISA §7.12.2)
// ---------------------------------------------------------------------------
__device__ __forceinline__ v16h load_a16(const half_t* A, int lda, int row, int k0,
                                         unsigned lane) {
  unsigned hs = (lane >> 4) & 1u;
  unsigned m  = lane & 15u;
  const half_t* base = A + (size_t)(row + (int)m) * lda + k0 + 8 * hs;
  union { v16h v; unsigned u[8]; } r;
#pragma unroll
  for (int j = 0; j < 4; ++j) r.u[j]     = *(const unsigned*)(base + 2 * j);
#pragma unroll
  for (int j = 0; j < 4; ++j) r.u[4 + j] = *(const unsigned*)(base + 16 + 2 * j);
  return r.v;
}

__device__ __forceinline__ v16h load_b16(const half_t* Wt, int ldw, int n0, int k0,
                                         unsigned lane) {
  unsigned hs = (lane >> 4) & 1u;
  unsigned n  = lane & 15u;
  const half_t* base = Wt + (size_t)(n0 + (int)n) * ldw + k0 + 8 * hs;
  union { v16h v; unsigned u[8]; } r;
#pragma unroll
  for (int j = 0; j < 4; ++j) r.u[j]     = *(const unsigned*)(base + 2 * j);
#pragma unroll
  for (int j = 0; j < 4; ++j) r.u[4 + j] = *(const unsigned*)(base + 16 + 2 * j);
  return r.v;
}

__device__ __forceinline__ v8f wmma_f16(v16h a, v16h b, v8f c) {
  return __builtin_amdgcn_wmma_f32_16x16x32_f16(false, a, false, b, (short)0, c,
                                                false, false);
}

__device__ __forceinline__ v8f vzero8() {
  v8f z;
#pragma unroll
  for (int i = 0; i < 8; ++i) z[i] = 0.0f;
  return z;
}

// ---------------------------------------------------------------------------
// Tensor Data Mover: DMA a 2D f16 tile [rows x cols] (row stride = ld_elems)
// from global memory into LDS at lds_byte_addr. D# layout per
// CDNA5 ISA. 2D: tile_dim2 = 0, groups 2/3 zeroed.
// 6-arg builtin form (clang-23 / therock-10.0 headers).
// ---------------------------------------------------------------------------
__device__ __forceinline__ void tdm_load_2d(unsigned lds_byte_addr,
                                            const half_t* gptr, int ld_elems,
                                            int rows, int cols) {
  unsigned long long ga = (unsigned long long)(size_t)gptr;
  v4u g0;
  g0[0] = 1u;                                           // count=1, user desc
  g0[1] = lds_byte_addr;                                // lds_addr
  g0[2] = (unsigned)(ga & 0xFFFFFFFFu);                 // global_addr[31:0]
  g0[3] = (unsigned)((ga >> 32) & 0x1FFFFFFu)           // global_addr[56:32]
        | (2u << 30);                                   // type=2 ("image")
  v8i g1;
  g1[0] = (int)(1u << 16);                              // data_size=1 (2 bytes)
  g1[1] = (int)((unsigned)cols << 16);                  // tensor_dim0 (lo16)
  g1[2] = (int)((unsigned)rows << 16);                  // tensor_dim1 (lo16)
  g1[3] = (int)((unsigned)cols << 16);                  // tile_dim0
  g1[4] = rows;                                         // tile_dim1 (tile_dim2=0)
  g1[5] = ld_elems;                                     // tensor_dim0_stride lo32
  g1[6] = 0;
  g1[7] = 0;
  v4i z4; z4[0] = 0; z4[1] = 0; z4[2] = 0; z4[3] = 0;
  v8i z8;
#pragma unroll
  for (int i = 0; i < 8; ++i) z8[i] = 0;
  __builtin_amdgcn_tensor_load_to_lds(g0, g1, z4, z4, z8, 0);
}

// ---------------------------------------------------------------------------
// Workgroup GEMM core: 8 waves, each owns a 16(M)x64(N) fp32 tile; the 64x32
// B tile is TDM-DMA'd into LDS (double-buffered) and shared by all waves.
// A fragments are software-pipelined one k-step ahead from global (L2-resident).
// All 4 B fragments are fetched before the WMMA chain so the DS waits are
// progressive (dscnt <= 6/4/2/0) instead of serializing each WMMA.
// ---------------------------------------------------------------------------
__device__ __forceinline__ void gemm_wg16x64(const half_t* A, int lda,
                                             const half_t* Wt, int ldw,
                                             int mbase, int nbase, int K,
                                             unsigned tid,
                                             half_t bs[2][64][32], v8f acc[4]) {
  unsigned lane = tid & 31u, w = tid >> 5;
  const half_t* wbase = Wt + (size_t)nbase * ldw;
  int cur = 0;
  if (w == 0) tdm_load_2d((unsigned)(size_t)&bs[0][0][0], wbase, ldw, 64, 32);

  v16h a = load_a16(A, lda, mbase, 0, lane);
  for (int k0 = 0; k0 < K; k0 += 32) {
    if (w == 0) __builtin_amdgcn_s_wait_tensorcnt(0);
    __syncthreads();   // publish bs[cur]; all reads of bs[cur^1] are done
    if (w == 0 && (k0 + 32 < K))
      tdm_load_2d((unsigned)(size_t)&bs[cur ^ 1][0][0], wbase + k0 + 32, ldw, 64, 32);
    v16h acur = a;
    if (k0 + 32 < K) a = load_a16(A, lda, mbase, k0 + 32, lane);  // pipeline A
    v16h bv[4];
#pragma unroll
    for (int t = 0; t < 4; ++t)
      bv[t] = load_b16(&bs[cur][0][0], 32, 16 * t, 0, lane);      // ds_load x8
#pragma unroll
    for (int t = 0; t < 4; ++t)
      acc[t] = wmma_f16(acur, bv[t], acc[t]);
    cur ^= 1;
  }
}

// ---------------------------------------------------------------------------
// Weight conversion: fp32 row-major [K,N] -> f16 K-contiguous [N,K]
// ---------------------------------------------------------------------------
__global__ void cvt_transpose(const float* __restrict__ src, half_t* __restrict__ dst,
                              int K, int N) {
  int idx = blockIdx.x * 256 + threadIdx.x;
  if (idx >= K * N) return;
  int k = idx % K;
  int n = idx / K;
  dst[idx] = (half_t)src[(size_t)k * N + n];
}

// wq/wk/wv: [H,1024,64] fp32 -> packed Wt[(which*1024 + h*64 + d)][k] f16
__global__ void cvt_qkv(const float* __restrict__ wq, const float* __restrict__ wk,
                        const float* __restrict__ wv, half_t* __restrict__ W) {
  int idx = blockIdx.x * 256 + threadIdx.x;
  if (idx >= 3 * 1024 * 1024) return;
  int k = idx & 1023;
  int n = (idx >> 10) & 1023;
  int which = idx >> 20;
  const float* src = (which == 0) ? wq : (which == 1) ? wk : wv;
  int hh = n >> 6, d = n & 63;
  W[idx] = (half_t)src[(size_t)hh * (1024 * 64) + (size_t)k * 64 + d];
}

// ---------------------------------------------------------------------------
// LayerNorm: one block per row of 1024, fp32 math, f16 output
// ---------------------------------------------------------------------------
__global__ void ln_kernel(const float* __restrict__ x, const float* __restrict__ g,
                          const float* __restrict__ be, half_t* __restrict__ out) {
  __shared__ float red[16];
  int row = blockIdx.x;
  const float* xr = x + (size_t)row * N_EMBD;
  float vals[4], s = 0.f, ss = 0.f;
#pragma unroll
  for (int i = 0; i < 4; ++i) {
    float v = xr[threadIdx.x + 256 * i];
    vals[i] = v; s += v; ss += v * v;
  }
#pragma unroll
  for (int m = 16; m >= 1; m >>= 1) {
    s  += __shfl_xor(s,  m, 32);
    ss += __shfl_xor(ss, m, 32);
  }
  int w = threadIdx.x >> 5;
  if ((threadIdx.x & 31) == 0) { red[w] = s; red[8 + w] = ss; }
  __syncthreads();
  float ts = 0.f, tss = 0.f;
#pragma unroll
  for (int i = 0; i < 8; ++i) { ts += red[i]; tss += red[8 + i]; }
  float mean = ts * (1.0f / N_EMBD);
  float var  = tss * (1.0f / N_EMBD) - mean * mean;
  float rstd = rsqrtf(var + 1e-5f);
#pragma unroll
  for (int i = 0; i < 4; ++i) {
    int c = threadIdx.x + 256 * i;
    out[(size_t)row * N_EMBD + c] = (half_t)((vals[i] - mean) * rstd * g[c] + be[c]);
  }
}

// ---------------------------------------------------------------------------
// Fused QKV projection: h1[8192,1024] @ Wqkv -> Q,K [B,H,T,64], Vt [B,H,64,T]
// ---------------------------------------------------------------------------
__global__ void gemm_qkv(const half_t* __restrict__ H1, const half_t* __restrict__ W,
                         half_t* __restrict__ Qo, half_t* __restrict__ Ko,
                         half_t* __restrict__ Vo) {
  __shared__ half_t bs[2][64][32];
  unsigned lane = threadIdx.x & 31u, w = threadIdx.x >> 5;
  unsigned hs = lane >> 4, ln = lane & 15u;
  int mbase = blockIdx.y * 128 + (int)w * 16;
  int nbase = blockIdx.x * 64;
  v8f acc[4];
#pragma unroll
  for (int t = 0; t < 4; ++t) acc[t] = vzero8();
  gemm_wg16x64(H1, N_EMBD, W, N_EMBD, mbase, nbase, N_EMBD, threadIdx.x, bs, acc);
#pragma unroll
  for (int t = 0; t < 4; ++t) {
    int n = nbase + 16 * t + (int)ln;       // 0..3071
    int which = n >> 10;
    int nn = n & 1023;
    int hh = nn >> 6, d = nn & 63;
#pragma unroll
    for (int r = 0; r < 8; ++r) {
      int m = mbase + r + 8 * (int)hs;      // 0..8191 = b*SEQ + t
      int b = m >> 11, tp = m & 2047;
      half_t val = (half_t)acc[t][r];
      size_t bh = (size_t)(b * N_HEAD + hh);
      if (which == 0)      Qo[(bh * SEQ + tp) * HEAD + d] = val;
      else if (which == 1) Ko[(bh * SEQ + tp) * HEAD + d] = val;
      else                 Vo[(bh * HEAD + d) * SEQ + tp] = val;
    }
  }
}

// ---------------------------------------------------------------------------
// Causal flash attention: grid (T/128, B*H), 8 waves; each wave = 16 q-rows.
// Online softmax in fp32; P transposed C-layout -> A-layout via per-wave LDS.
// ---------------------------------------------------------------------------
__global__ void attn_kernel(const half_t* __restrict__ Q, const half_t* __restrict__ Kb,
                            const half_t* __restrict__ Vt, half_t* __restrict__ O) {
  __shared__ half_t plds[8][16][32];
  unsigned tid = threadIdx.x;
  unsigned w = tid >> 5, lane = tid & 31u;
  unsigned hs = lane >> 4, ln = lane & 15u;
  int bh = blockIdx.y;
  int qbase = blockIdx.x * 128 + (int)w * 16;
  const half_t* Qh = Q  + (size_t)bh * SEQ * HEAD;
  const half_t* Kh = Kb + (size_t)bh * SEQ * HEAD;
  const half_t* Vh = Vt + (size_t)bh * HEAD * SEQ;

  v16h aq0 = load_a16(Qh, HEAD, qbase, 0, lane);
  v16h aq1 = load_a16(Qh, HEAD, qbase, 32, lane);

  v8f accO[4];
#pragma unroll
  for (int t = 0; t < 4; ++t) accO[t] = vzero8();
  float mrow[8], srow[8];
#pragma unroll
  for (int r = 0; r < 8; ++r) { mrow[r] = NEGINF; srow[r] = 0.0f; }

  const float scale = 0.125f;  // 1/sqrt(64)
  int qend = qbase + 15;

  for (int s0 = 0; s0 <= qend; s0 += 32) {
    // fetch all 4 K fragments before the S-WMMA chain
    v16h bk[4];
    bk[0] = load_b16(Kh, HEAD, s0,      0,  lane);
    bk[1] = load_b16(Kh, HEAD, s0,      32, lane);
    bk[2] = load_b16(Kh, HEAD, s0 + 16, 0,  lane);
    bk[3] = load_b16(Kh, HEAD, s0 + 16, 32, lane);
    v8f S0 = vzero8(), S1 = vzero8();
    S0 = wmma_f16(aq0, bk[0], S0);
    S0 = wmma_f16(aq1, bk[1], S0);
    S1 = wmma_f16(aq0, bk[2], S1);
    S1 = wmma_f16(aq1, bk[3], S1);

    // fetch all 4 V fragments early (consumed after the softmax VALU block)
    v16h bv[4];
#pragma unroll
    for (int t = 0; t < 4; ++t)
      bv[t] = load_b16(Vh, SEQ, t * 16, s0, lane);  // Vt rows are d, cols are s

    int col0 = s0 + (int)ln, col1 = s0 + 16 + (int)ln;
#pragma unroll
    for (int r = 0; r < 8; ++r) {
      int qrow = qbase + r + 8 * (int)hs;
      float v0 = (col0 <= qrow) ? S0[r] * scale : NEGINF;
      float v1 = (col1 <= qrow) ? S1[r] * scale : NEGINF;
      float mx = fmaxf(v0, v1);
#pragma unroll
      for (int msk = 8; msk >= 1; msk >>= 1) mx = fmaxf(mx, __shfl_xor(mx, msk, 32));
      float mnew = fmaxf(mrow[r], mx);
      float corr = __expf(mrow[r] - mnew);
      float p0 = __expf(v0 - mnew);
      float p1 = __expf(v1 - mnew);
      float ps = p0 + p1;
#pragma unroll
      for (int msk = 8; msk >= 1; msk >>= 1) ps += __shfl_xor(ps, msk, 32);
      srow[r] = srow[r] * corr + ps;
      mrow[r] = mnew;
#pragma unroll
      for (int t = 0; t < 4; ++t) accO[t][r] *= corr;
      plds[w][r + 8 * hs][ln]      = (half_t)p0;
      plds[w][r + 8 * hs][16 + ln] = (half_t)p1;
    }
    // Per-wave LDS region; DS ops are in-order within a wave -> no barrier.
    v16h pa = load_a16(&plds[w][0][0], 32, 0, 0, lane);
#pragma unroll
    for (int t = 0; t < 4; ++t)
      accO[t] = wmma_f16(pa, bv[t], accO[t]);
  }

  int b = bh >> 4, h = bh & 15;
#pragma unroll
  for (int r = 0; r < 8; ++r) {
    float inv = 1.0f / srow[r];
#pragma unroll
    for (int t = 0; t < 4; ++t) {
      int m = qbase + r + 8 * (int)hs;
      int c = h * HEAD + t * 16 + (int)ln;
      O[(size_t)(b * SEQ + m) * N_EMBD + c] = (half_t)(accO[t][r] * inv);
    }
  }
}

// ---------------------------------------------------------------------------
// Output projection + residual:  out = x + (attn_o @ Wo + bo)   (fp32 out)
// ---------------------------------------------------------------------------
__global__ void gemm_wo(const half_t* __restrict__ Ao, const half_t* __restrict__ W,
                        const float* __restrict__ bias, const float* __restrict__ x,
                        float* __restrict__ out) {
  __shared__ half_t bs[2][64][32];
  unsigned lane = threadIdx.x & 31u, w = threadIdx.x >> 5;
  unsigned hs = lane >> 4, ln = lane & 15u;
  int mbase = blockIdx.y * 128 + (int)w * 16;
  int nbase = blockIdx.x * 64;
  v8f acc[4];
#pragma unroll
  for (int t = 0; t < 4; ++t) acc[t] = vzero8();
  gemm_wg16x64(Ao, N_EMBD, W, N_EMBD, mbase, nbase, N_EMBD, threadIdx.x, bs, acc);
#pragma unroll
  for (int t = 0; t < 4; ++t) {
    int n = nbase + 16 * t + (int)ln;
    float bn = bias[n];
#pragma unroll
    for (int r = 0; r < 8; ++r) {
      int m = mbase + r + 8 * (int)hs;
      size_t idx = (size_t)m * N_EMBD + n;
      out[idx] = x[idx] + acc[t][r] + bn;
    }
  }
}

// FFN layer 1: relu(h2 @ W1 + b1) -> f16 [8192, 4096]
__global__ void gemm_ffn1(const half_t* __restrict__ H2, const half_t* __restrict__ W,
                          const float* __restrict__ bias, half_t* __restrict__ out) {
  __shared__ half_t bs[2][64][32];
  unsigned lane = threadIdx.x & 31u, w = threadIdx.x >> 5;
  unsigned hs = lane >> 4, ln = lane & 15u;
  int mbase = blockIdx.y * 128 + (int)w * 16;
  int nbase = blockIdx.x * 64;
  v8f acc[4];
#pragma unroll
  for (int t = 0; t < 4; ++t) acc[t] = vzero8();
  gemm_wg16x64(H2, N_EMBD, W, N_EMBD, mbase, nbase, N_EMBD, threadIdx.x, bs, acc);
#pragma unroll
  for (int t = 0; t < 4; ++t) {
    int n = nbase + 16 * t + (int)ln;
    float bn = bias[n];
#pragma unroll
    for (int r = 0; r < 8; ++r) {
      int m = mbase + r + 8 * (int)hs;
      out[(size_t)m * FFN + n] = (half_t)fmaxf(acc[t][r] + bn, 0.0f);
    }
  }
}

// FFN layer 2 + residual accumulate into d_out:  out += ff @ W2 + b2
__global__ void gemm_ffn2(const half_t* __restrict__ F, const half_t* __restrict__ W,
                          const float* __restrict__ bias, float* __restrict__ out) {
  __shared__ half_t bs[2][64][32];
  unsigned lane = threadIdx.x & 31u, w = threadIdx.x >> 5;
  unsigned hs = lane >> 4, ln = lane & 15u;
  int mbase = blockIdx.y * 128 + (int)w * 16;
  int nbase = blockIdx.x * 64;
  v8f acc[4];
#pragma unroll
  for (int t = 0; t < 4; ++t) acc[t] = vzero8();
  gemm_wg16x64(F, FFN, W, FFN, mbase, nbase, FFN, threadIdx.x, bs, acc);
#pragma unroll
  for (int t = 0; t < 4; ++t) {
    int n = nbase + 16 * t + (int)ln;
    float bn = bias[n];
#pragma unroll
    for (int r = 0; r < 8; ++r) {
      int m = mbase + r + 8 * (int)hs;
      size_t idx = (size_t)m * N_EMBD + n;
      out[idx] = out[idx] + acc[t][r] + bn;
    }
  }
}

// ---------------------------------------------------------------------------
extern "C" void kernel_launch(void* const* d_in, const int* in_sizes, int n_in,
                              void* d_out, int out_size, void* d_ws, size_t ws_size,
                              hipStream_t stream) {
  const float* x   = (const float*)d_in[0];
  const float* wq  = (const float*)d_in[1];
  const float* wk  = (const float*)d_in[2];
  const float* wv  = (const float*)d_in[3];
  const float* wo  = (const float*)d_in[4];
  const float* bo  = (const float*)d_in[5];
  const float* w1  = (const float*)d_in[6];
  const float* b1  = (const float*)d_in[7];
  const float* w2  = (const float*)d_in[8];
  const float* b2  = (const float*)d_in[9];
  const float* g1  = (const float*)d_in[10];
  const float* be1 = (const float*)d_in[11];
  const float* g2  = (const float*)d_in[12];
  const float* be2 = (const float*)d_in[13];
  float* out = (float*)d_out;

  const size_t MB = 1024ull * 1024ull;
  char* ws = (char*)d_ws;
  half_t* WtQKV = (half_t*)(ws + 0 * MB);    // 6 MB  [3072][1024] f16
  half_t* WtO   = (half_t*)(ws + 6 * MB);    // 2 MB  [1024][1024]
  half_t* Wt1   = (half_t*)(ws + 8 * MB);    // 8 MB  [4096][1024]
  half_t* Wt2   = (half_t*)(ws + 16 * MB);   // 8 MB  [1024][4096]
  half_t* H1    = (half_t*)(ws + 24 * MB);   // 16 MB h1, reused as h2
  half_t* Qb    = (half_t*)(ws + 40 * MB);   // 16 MB [B,H,T,64]
  half_t* Kbf   = (half_t*)(ws + 56 * MB);   // 16 MB [B,H,T,64]
  half_t* Vtb   = (half_t*)(ws + 72 * MB);   // 16 MB [B,H,64,T]
  half_t* AO    = (half_t*)(ws + 88 * MB);   // 16 MB attn out [B*T, D]
  half_t* FF    = (half_t*)(ws + 40 * MB);   // 64 MB, aliases Qb..AO (dead by then)

  cvt_qkv<<<(3 * 1024 * 1024) / 256, 256, 0, stream>>>(wq, wk, wv, WtQKV);
  cvt_transpose<<<(1024 * 1024) / 256, 256, 0, stream>>>(wo, WtO, 1024, 1024);
  cvt_transpose<<<(1024 * 4096) / 256, 256, 0, stream>>>(w1, Wt1, 1024, 4096);
  cvt_transpose<<<(4096 * 1024) / 256, 256, 0, stream>>>(w2, Wt2, 4096, 1024);

  ln_kernel<<<ROWS, 256, 0, stream>>>(x, g1, be1, H1);
  gemm_qkv<<<dim3(48, 64), 256, 0, stream>>>(H1, WtQKV, Qb, Kbf, Vtb);
  attn_kernel<<<dim3(SEQ / 128, BATCH * N_HEAD), 256, 0, stream>>>(Qb, Kbf, Vtb, AO);
  gemm_wo<<<dim3(16, 64), 256, 0, stream>>>(AO, WtO, bo, x, out);
  ln_kernel<<<ROWS, 256, 0, stream>>>(out, g2, be2, H1);
  gemm_ffn1<<<dim3(64, 64), 256, 0, stream>>>(H1, Wt1, b1, FF);
  gemm_ffn2<<<dim3(16, 64), 256, 0, stream>>>(FF, Wt2, b2, out);
}
